// Graph2PCGNN_8031588843786
// MI455X (gfx1250) — compile-verified, matrix operand order checked by reading, not run on hardware
//
#include <hip/hip_runtime.h>
#include <hip/hip_bf16.h>

typedef __attribute__((ext_vector_type(16))) __bf16 v16bf;
typedef __attribute__((ext_vector_type(8)))  __bf16 v8bf;
typedef __attribute__((ext_vector_type(8)))  float  v8f;

#define N_NODES  50000
#define N_EDGES  800000
#define D_FEAT   128
#define PHI_HID  256
#define MLP_HID  256
#define N_CLASSES 2
#define N_GRAPHS 128
#define N_REP    64   // replicas of the pooled accumulator (contention relief, L2-resident)
#define MT_GROUP 5    // m-tiles per wave in node GEMM (3125 = 625*5, exact)
#define M_TILES  (N_NODES / 16)   // 3125

// K-index mapping for 16-bit WMMA A/B operands (wave32):
// VGPR pair v holds two consecutive K values; lanes 0-15 cover K {0..7,16..23},
// lanes 16-31 cover K {8..15,24..31} (per CDNA5 ISA 7.12.2 16-bit A-matrix table).
__device__ __forceinline__ int kpair_base(int v, int half) {
    int b = (v < 4) ? (2 * v) : (16 + 2 * (v - 4));
    return b + (half ? 8 : 0);
}

// ---------------------------------------------------------------------------
// Pre-pack phi_w1 -> bf16 WMMA B-operand layout.
// wpack[((ntile*4 + kstep)*32 + lane)*16 + e]; 32x4x32x16 bf16 = 128 KB.
// ---------------------------------------------------------------------------
__global__ void __launch_bounds__(256)
k_pack_w1(const float* __restrict__ W1, __bf16* __restrict__ wpack)
{
    const int t = blockIdx.x * blockDim.x + threadIdx.x;   // 0..4095
    if (t >= 32 * 4 * 32) return;
    const int lane  = t & 31;
    const int ks    = (t >> 5) & 3;
    const int ntile = t >> 7;
    const int half  = lane >> 4;
    const int wslice = ntile >> 4;                   // 0 -> W1a rows, 1 -> W1b rows
    const int wcol   = ((ntile & 15) << 4) + (lane & 15);
    __bf16* dst = wpack + (size_t)t * 16;
    #pragma unroll
    for (int v = 0; v < 8; ++v) {
        int k = ks * 32 + kpair_base(v, half);
        dst[2 * v]     = (__bf16)W1[(size_t)(wslice * 128 + k)     * PHI_HID + wcol];
        dst[2 * v + 1] = (__bf16)W1[(size_t)(wslice * 128 + k + 1) * PHI_HID + wcol];
    }
}

// ---------------------------------------------------------------------------
// Pre-pack X -> bf16 WMMA A-operand layout (shared by all 32 n-tiles).
// xpack[((mt*4 + kstep)*32 + lane)*16 + e]; 3125x4x32x16 bf16 = 12.8 MB.
// ---------------------------------------------------------------------------
__global__ void __launch_bounds__(256)
k_pack_x(const float* __restrict__ X, __bf16* __restrict__ xpack)
{
    const int t = blockIdx.x * blockDim.x + threadIdx.x;   // 0..399999
    if (t >= M_TILES * 4 * 32) return;
    const int lane = t & 31;
    const int ks   = (t >> 5) & 3;
    const int mt   = t >> 7;
    const int half = lane >> 4;
    const int mrow = mt * 16 + (lane & 15);
    const float* xrow = X + (size_t)mrow * D_FEAT;
    v16bf a;
    #pragma unroll
    for (int v = 0; v < 8; ++v) {
        int k = ks * 32 + kpair_base(v, half);
        float2 f = *(const float2*)(xrow + k);
        a[2 * v]     = (__bf16)f.x;
        a[2 * v + 1] = (__bf16)f.y;
    }
    *reinterpret_cast<v16bf*>(xpack + (size_t)t * 16) = a;
}

// ---------------------------------------------------------------------------
// Stage A: AB[n, 0:256] = X[n] @ phi_w1[0:128,:],  AB[n, 256:512] = X[n] @ phi_w1[128:256,:]
// Pure WMMA loop, software-pipelined: while the 4 WMMAs of m-tile i execute,
// the A operands of m-tile i+1 are already in flight (ping-pong registers).
// Each wave: 5 m-tiles x 1 n-tile, B register-resident across the group.
// Waves: 625 m-groups x 32 ntiles = 20000 -> 2500 blocks of 8 waves.
// ---------------------------------------------------------------------------
__global__ void __launch_bounds__(256)
k_node_gemm(const __bf16* __restrict__ xpack, const __bf16* __restrict__ wpack,
            __bf16* __restrict__ AB)
{
    const int lane = threadIdx.x & 31;
    const int half = lane >> 4;
    const int l15  = lane & 15;
    const int waveId = blockIdx.x * 8 + (threadIdx.x >> 5);
    const int mg    = waveId >> 5;       // m-group (0..624)
    const int ntile = waveId & 31;       // tile col over the 512-wide AB
    if (mg >= N_NODES / (16 * MT_GROUP)) return;

    // B operands: one aligned 32B vector load per k-step, held in registers
    v16bf bmat[4];
    #pragma unroll
    for (int ks = 0; ks < 4; ++ks)
        bmat[ks] = *reinterpret_cast<const v16bf*>(
            wpack + ((size_t)(ntile * 4 + ks) * 32 + lane) * 16);

    const int outcol = (ntile << 4) + l15;
    const __bf16* agrp = xpack + ((size_t)(mg * MT_GROUP) * 4 * 32 + lane) * 16;

    // prologue: load all 4 k-step A operands of the first m-tile
    v16bf apipe[2][4];
    #pragma unroll
    for (int ks = 0; ks < 4; ++ks)
        apipe[0][ks] = *reinterpret_cast<const v16bf*>(agrp + (size_t)ks * 32 * 16);

    #pragma unroll
    for (int i = 0; i < MT_GROUP; ++i) {
        const int cur = i & 1;
        // issue next m-tile's A loads before consuming the current operands
        if (i + 1 < MT_GROUP) {
            const __bf16* anext = agrp + (size_t)(i + 1) * 4 * 32 * 16;
            #pragma unroll
            for (int ks = 0; ks < 4; ++ks)
                apipe[cur ^ 1][ks] =
                    *reinterpret_cast<const v16bf*>(anext + (size_t)ks * 32 * 16);
        }

        v8f acc = {};
        #pragma unroll
        for (int ks = 0; ks < 4; ++ks)
            acc = __builtin_amdgcn_wmma_f32_16x16x32_bf16(false, apipe[cur][ks], false,
                                                          bmat[ks], (short)0, acc,
                                                          false, false);

        // C/D layout: VGPR r -> row r (lanes 0-15) / row r+8 (lanes 16-31), col = l15
        const int mt = mg * MT_GROUP + i;
        #pragma unroll
        for (int r = 0; r < 8; ++r) {
            int m = mt * 16 + r + (half ? 8 : 0);
            AB[(size_t)m * 512 + outcol] = (__bf16)acc[r];
        }
    }
}

// ---------------------------------------------------------------------------
// Generic small-GEMM with WMMA: C = maybe_relu(A@B + bias[n] + svec[m]*avec[n])
// A: [M,K] f32 row-major, B: [K,N] f32 row-major, C: [M,N] f32. K % 32 == 0.
// One 16x16 tile per wave, 8 waves per block. Used only for tiny epilogue GEMMs.
// ---------------------------------------------------------------------------
__global__ void __launch_bounds__(256)
k_wmma_gemm(const float* __restrict__ A, const float* __restrict__ B,
            float* __restrict__ C, int M, int N, int K,
            const float* __restrict__ bias, const float* __restrict__ svec,
            const float* __restrict__ avec, int relu)
{
    const int lane = threadIdx.x & 31;
    const int half = lane >> 4;
    const int l15  = lane & 15;
    const int tilesPerRow = N >> 4;
    const int waveTile = blockIdx.x * 8 + (threadIdx.x >> 5);
    const int mt = waveTile / tilesPerRow;
    const int nt = waveTile % tilesPerRow;
    if (mt * 16 >= M) return;    // wave-uniform: EXEC stays all-ones for WMMA

    const int nc   = nt * 16 + l15;
    const int mrow = mt * 16 + l15;
    const float* arow = A + (size_t)mrow * K;
    const float* bcol = B + nc;

    v8f acc = {};
    for (int k0 = 0; k0 < K; k0 += 32) {
        v16bf a, b;
        #pragma unroll
        for (int v = 0; v < 8; ++v) {
            int k = k0 + kpair_base(v, half);
            float2 f = *(const float2*)(arow + k);
            a[2 * v]     = (__bf16)f.x;
            a[2 * v + 1] = (__bf16)f.y;
            b[2 * v]     = (__bf16)bcol[(size_t)k       * N];
            b[2 * v + 1] = (__bf16)bcol[(size_t)(k + 1) * N];
        }
        acc = __builtin_amdgcn_wmma_f32_16x16x32_bf16(false, a, false, b,
                                                      (short)0, acc, false, false);
    }
    const float badd = bias ? bias[nc] : 0.0f;
    const float aadd = avec ? avec[nc] : 0.0f;
    #pragma unroll
    for (int r = 0; r < 8; ++r) {
        int m = mt * 16 + r + (half ? 8 : 0);
        float v = acc[r] + badd;
        if (svec) v += svec[m] * aadd;
        if (relu) v = fmaxf(v, 0.0f);
        C[(size_t)m * N + nc] = v;
    }
}

// ---------------------------------------------------------------------------
// Edge phase: for each edge e, g = batch[row[e]];
//   pooled_rep[rep][g][:] += relu(A[row[e]] + B[col[e]] + phi_b1)
// One edge per wave-iteration: lane l handles features [8l, 8l+8).
// A/B rows are bf16 (L2-resident, 51 MB); accumulators f32 via global atomics.
// ---------------------------------------------------------------------------
__global__ void __launch_bounds__(256)
k_edges(const int* __restrict__ row, const int* __restrict__ col,
        const int* __restrict__ batch, const __bf16* __restrict__ AB,
        const float* __restrict__ phi_b1,
        float* __restrict__ rep, float* __restrict__ cnt_rep)
{
    const int lane = threadIdx.x & 31;
    const int waveGlobal = (blockIdx.x * blockDim.x + threadIdx.x) >> 5;
    const int nWaves = (gridDim.x * blockDim.x) >> 5;
    const int repIdx = blockIdx.x & (N_REP - 1);
    float* myRep = rep     + (size_t)repIdx * (N_GRAPHS * PHI_HID);
    float* myCnt = cnt_rep + (size_t)repIdx * N_GRAPHS;

    float bl[8];
    #pragma unroll
    for (int j = 0; j < 8; ++j) bl[j] = phi_b1[lane * 8 + j];

    for (int e = waveGlobal; e < N_EDGES; e += nWaves) {
        const int r = row[e];
        const int c = col[e];
        const int g = batch[r];

        // prefetch next iteration's gather rows into cache (global_prefetch_b8)
        const int e2 = e + nWaves;
        if (e2 < N_EDGES) {
            __builtin_prefetch(AB + (size_t)row[e2] * 512 + (lane << 3), 0, 0);
            __builtin_prefetch(AB + (size_t)col[e2] * 512 + 256 + (lane << 3), 0, 0);
        }

        const v8bf av = *reinterpret_cast<const v8bf*>(AB + (size_t)r * 512 + (lane << 3));
        const v8bf bv = *reinterpret_cast<const v8bf*>(AB + (size_t)c * 512 + 256 + (lane << 3));
        float* dst = myRep + g * PHI_HID + (lane << 3);
        #pragma unroll
        for (int j = 0; j < 8; ++j) {
            float v = (float)av[j] + (float)bv[j] + bl[j];
            v = fmaxf(v, 0.0f);
            atomicAdd(dst + j, v);
        }
        if (lane == 0) atomicAdd(myCnt + g, 1.0f);
    }
}

// ---------------------------------------------------------------------------
__global__ void k_zero(float* __restrict__ p, size_t n)
{
    size_t i = (size_t)blockIdx.x * blockDim.x + threadIdx.x;
    const size_t stride = (size_t)gridDim.x * blockDim.x;
    for (; i < n; i += stride) p[i] = 0.0f;
}

__global__ void k_reduce(const float* __restrict__ rep, const float* __restrict__ cnt_rep,
                         float* __restrict__ pooled_h, float* __restrict__ cnt)
{
    const int i = blockIdx.x * blockDim.x + threadIdx.x;
    if (i < N_GRAPHS * PHI_HID) {
        float s = 0.0f;
        for (int r = 0; r < N_REP; ++r) s += rep[(size_t)r * N_GRAPHS * PHI_HID + i];
        pooled_h[i] = s;
    }
    if (i < N_GRAPHS) {
        float s = 0.0f;
        for (int r = 0; r < N_REP; ++r) s += cnt_rep[r * N_GRAPHS + i];
        cnt[i] = s;
    }
}

// out[g,c] = sum_k h1[g,k]*mlp_w2[k,c] + mlp_b2[c]   (N=2, trivial)
__global__ void k_readout(const float* __restrict__ h1, const float* __restrict__ w2,
                          const float* __restrict__ b2, float* __restrict__ out)
{
    const int t = threadIdx.x;                 // 256 threads = 128 graphs x 2 classes
    const int g = t >> 1, c = t & 1;
    float s = b2[c];
    for (int k = 0; k < MLP_HID; ++k) s += h1[g * MLP_HID + k] * w2[k * N_CLASSES + c];
    out[g * N_CLASSES + c] = s;
}

// ---------------------------------------------------------------------------
extern "C" void kernel_launch(void* const* d_in, const int* in_sizes, int n_in,
                              void* d_out, int out_size, void* d_ws, size_t ws_size,
                              hipStream_t stream)
{
    const float* x       = (const float*)d_in[0];
    const int*   row     = (const int*)d_in[1];            // edge_index[0]
    const int*   col     = row + N_EDGES;                  // edge_index[1]
    const int*   batch   = (const int*)d_in[2];
    const float* phi_w1  = (const float*)d_in[3];
    const float* phi_b1  = (const float*)d_in[4];
    const float* phi_w2  = (const float*)d_in[5];
    const float* phi_b2  = (const float*)d_in[6];
    const float* mlp_w1  = (const float*)d_in[7];
    const float* mlp_b1  = (const float*)d_in[8];
    const float* mlp_w2  = (const float*)d_in[9];
    const float* mlp_b2  = (const float*)d_in[10];
    float* out = (float*)d_out;

    // workspace layout (bytes); total ~69.5 MiB
    char* ws = (char*)d_ws;
    __bf16* AB       = (__bf16*)(ws);                                  // 50000*512 bf16 = 51,200,000 B
    size_t off       = (size_t)N_NODES * 512 * sizeof(__bf16);
    float* rep       = (float*)(ws + off);                             // 64*128*256 f32
    off += (size_t)N_REP * N_GRAPHS * PHI_HID * sizeof(float);
    float* cnt_rep   = (float*)(ws + off);                             // 64*128 f32
    off += (size_t)N_REP * N_GRAPHS * sizeof(float);
    float* pooled_h  = (float*)(ws + off);                             // 128*256 f32
    off += (size_t)N_GRAPHS * PHI_HID * sizeof(float);
    float* cnt       = (float*)(ws + off);                             // 128 f32
    off += 128 * sizeof(float);
    float* pooled    = (float*)(ws + off);                             // 128*128 f32
    off += (size_t)N_GRAPHS * D_FEAT * sizeof(float);
    float* h1        = (float*)(ws + off);                             // 128*256 f32
    off += (size_t)N_GRAPHS * MLP_HID * sizeof(float);
    __bf16* wpack    = (__bf16*)(ws + off);                            // 131,072 B
    off += (size_t)32 * 4 * 32 * 16 * sizeof(__bf16);
    __bf16* xpack    = (__bf16*)(ws + off);                            // 3125*4*32*16 bf16 = 12.8 MB

    // 1) zero the replicated accumulators (rep + cnt_rep are contiguous)
    const size_t nzero = (size_t)N_REP * (N_GRAPHS * PHI_HID + N_GRAPHS);
    k_zero<<<1024, 256, 0, stream>>>(rep, nzero);

    // 2) pre-pack phi_w1 and X into bf16 WMMA operand layouts
    k_pack_w1<<<16, 256, 0, stream>>>(phi_w1, wpack);
    k_pack_x<<<(M_TILES * 4 * 32 + 255) / 256, 256, 0, stream>>>(x, xpack);

    // 3) A|B = X @ [W1a | W1b]  (pure bf16 WMMA loop, software-pipelined)
    k_node_gemm<<<(N_NODES / (16 * MT_GROUP)) * 32 / 8, 256, 0, stream>>>(xpack, wpack, AB);

    // 4) edge phase: relu(A[row]+B[col]+b1) -> replicated per-graph accumulators
    k_edges<<<512, 256, 0, stream>>>(row, col, batch, AB, phi_b1, rep, cnt_rep);

    // 5) reduce replicas
    k_reduce<<<(N_GRAPHS * PHI_HID + 255) / 256, 256, 0, stream>>>(rep, cnt_rep, pooled_h, cnt);

    // 6) pooled = pooled_h @ phi_w2 + cnt[g]*phi_b2   [128,256]@[256,128]
    k_wmma_gemm<<<(N_GRAPHS / 16) * (D_FEAT / 16) / 8, 256, 0, stream>>>(
        pooled_h, phi_w2, pooled, N_GRAPHS, D_FEAT, PHI_HID,
        nullptr, cnt, phi_b2, 0);

    // 7) h1 = relu(pooled @ mlp_w1 + mlp_b1)          [128,128]@[128,256]
    k_wmma_gemm<<<(N_GRAPHS / 16) * (MLP_HID / 16) / 8, 256, 0, stream>>>(
        pooled, mlp_w1, h1, N_GRAPHS, MLP_HID, D_FEAT,
        mlp_b1, nullptr, nullptr, 1);

    // 8) out = h1 @ mlp_w2 + mlp_b2                   [128,256]@[256,2]
    k_readout<<<1, 256, 0, stream>>>(h1, mlp_w2, mlp_b2, out);
}
